// CrossAttentionLayer_8486855376916
// MI455X (gfx1250) — matrix operand order repaired and checked by reading, not compiled, not run on hardware
//
#include <hip/hip_runtime.h>

#define B_   4
#define N_   2048
#define M_   2048
#define DIN_ 512
#define H_   8
#define DK_  64
#define DV_  64

typedef __attribute__((ext_vector_type(16))) __bf16 v16bf;
typedef __attribute__((ext_vector_type(8)))  float  v8f;

#if defined(__AMDGCN__) && __has_builtin(__builtin_amdgcn_global_load_async_to_lds_b128)
#define HAVE_ASYNC_LDS 1
#else
#define HAVE_ASYNC_LDS 0
#endif

union FragBF {
  v16bf v;
  int4  q[2];
  unsigned short u[16];
};

__device__ __forceinline__ unsigned short f32_to_bf16(float f) {
  unsigned u = __float_as_uint(f);
  u += 0x7fffu + ((u >> 16) & 1u);   // round-to-nearest-even
  return (unsigned short)(u >> 16);
}

// pack two fp32 -> packed bf16x2 (RNE), using v_perm_b32 to grab high halves
__device__ __forceinline__ unsigned pack_bf16x2(float a, float b) {
  unsigned ua = __float_as_uint(a); ua += 0x7fffu + ((ua >> 16) & 1u);
  unsigned ub = __float_as_uint(b); ub += 0x7fffu + ((ub >> 16) & 1u);
  return __builtin_amdgcn_perm(ub, ua, 0x07060302);  // {ub.hi16, ua.hi16}
}

__device__ __forceinline__ v8f wmma_bf16(const FragBF& a, const FragBF& b, v8f c) {
  return __builtin_amdgcn_wmma_f32_16x16x32_bf16(
      false, a.v, false, b.v, (short)0, c, false, false);
}

// 16-byte global -> LDS copy: async DMA (ASYNCcnt) when available
__device__ __forceinline__ void g2l_16(const void* g, void* l) {
#if HAVE_ASYNC_LDS
  typedef int v4i_gv __attribute__((vector_size(16)));
  typedef __attribute__((address_space(1))) v4i_gv* gp_t;
  typedef __attribute__((address_space(3))) v4i_gv* lp_t;
  __builtin_amdgcn_global_load_async_to_lds_b128(
      (gp_t)(size_t)g, (lp_t)(unsigned)(size_t)l, 0, 0);
#else
  *(int4*)l = *(const int4*)g;
#endif
}

__device__ __forceinline__ void wait_async_lds() {
#if HAVE_ASYNC_LDS
#if __has_builtin(__builtin_amdgcn_s_wait_asynccnt)
  __builtin_amdgcn_s_wait_asynccnt(0);
#else
  asm volatile("s_wait_asynccnt 0x0" ::: "memory");
#endif
#endif
}

// ---------------------------------------------------------------------------
// Kernel 1: weight fp32 (in,out) -> bf16 transposed (out,in), 32x32 LDS tiles
// ---------------------------------------------------------------------------
__global__ __launch_bounds__(256) void wt_cvt_kernel(
    const float* __restrict__ Wq, const float* __restrict__ Wk,
    const float* __restrict__ Wv,
    unsigned short* __restrict__ WTq, unsigned short* __restrict__ WTk,
    unsigned short* __restrict__ WTv) {
  __shared__ unsigned short tile[32][33];
  const int bid = blockIdx.x;               // 3 * 16 * 16 blocks
  const int mat = bid >> 8;
  const int rem = bid & 255;
  const int to = (rem >> 4) * 32;           // out-col tile base
  const int tc = (rem & 15) * 32;           // in-row  tile base
  const float* W = (mat == 0) ? Wq : (mat == 1) ? Wk : Wv;
  unsigned short* WT = (mat == 0) ? WTq : (mat == 1) ? WTk : WTv;
  const int col = threadIdx.x & 31;
  const int row = threadIdx.x >> 5;         // 0..7
#pragma unroll
  for (int j = 0; j < 4; ++j) {             // coalesced read along out-dim
    const int cl = row + j * 8;
    tile[col][cl] = f32_to_bf16(W[(size_t)(tc + cl) * DIN_ + to + col]);
  }
  __syncthreads();
#pragma unroll
  for (int j = 0; j < 4; ++j) {             // coalesced write along in-dim
    const int ol = row + j * 8;
    WT[(size_t)(to + ol) * DIN_ + tc + col] = tile[ol][col];
  }
}

// ---------------------------------------------------------------------------
// Kernel 2: projection  y = x @ W + b.  Block = 16 rows; wave w = 64-col head.
// A tile (16x32) staged once per block in LDS as bf16.
// ---------------------------------------------------------------------------
__global__ __launch_bounds__(256) void proj_kernel(
    const float* __restrict__ x,             // (B*rows, 512) fp32
    const unsigned short* __restrict__ WT,   // (512 out, 512 in) bf16
    const float* __restrict__ bias,          // (512,)
    unsigned short* __restrict__ out,
    int transposed) {
  __shared__ alignas(16) unsigned short Xlds[16 * 32];

  const int lane = threadIdx.x & 31;
  const int wave = threadIdx.x >> 5;
  const int half = lane >> 4;
  const int l15  = lane & 15;
  const int row0 = blockIdx.x * 16;
  const int col0 = wave * 64;

  const int trow = threadIdx.x >> 4;         // 0..15 (stager row)
  const int tcol = (threadIdx.x & 15) * 2;   // 0,2,...,30

  v8f acc[4] = {};

  for (int c0 = 0; c0 < DIN_; c0 += 32) {
    // ---- stage A tile: 2 fp32 per thread -> packed bf16x2 in LDS ----
    {
      const float* xp = x + (size_t)(row0 + trow) * DIN_ + c0 + tcol;
      float2 f = *(const float2*)xp;
      *(unsigned*)&Xlds[trow * 32 + tcol] = pack_bf16x2(f.x, f.y);
      if (c0 + 32 < DIN_) __builtin_prefetch(xp + 32, 0, 1);
    }
    __syncthreads();

    // ---- A fragment from LDS (2 x ds_load_b128) ----
    FragBF a;
    {
      const unsigned short* ap = &Xlds[l15 * 32];
      a.q[0] = *(const int4*)(ap + half * 8);
      a.q[1] = *(const int4*)(ap + half * 8 + 16);
    }
    // ---- all 4 B fragments first (one clause of 8 b128), then 4 WMMAs ----
    FragBF b0, b1, b2, b3;
    {
      const size_t wbase = (size_t)(col0 + l15) * DIN_ + c0 + half * 8;
      const int4* wp0 = (const int4*)(WT + wbase);
      const int4* wp1 = (const int4*)(WT + wbase + 16 * DIN_);
      const int4* wp2 = (const int4*)(WT + wbase + 32 * DIN_);
      const int4* wp3 = (const int4*)(WT + wbase + 48 * DIN_);
      b0.q[0] = wp0[0]; b0.q[1] = wp0[2];
      b1.q[0] = wp1[0]; b1.q[1] = wp1[2];
      b2.q[0] = wp2[0]; b2.q[1] = wp2[2];
      b3.q[0] = wp3[0]; b3.q[1] = wp3[2];
    }
    acc[0] = wmma_bf16(a, b0, acc[0]);
    acc[1] = wmma_bf16(a, b1, acc[1]);
    acc[2] = wmma_bf16(a, b2, acc[2]);
    acc[3] = wmma_bf16(a, b3, acc[3]);
    __syncthreads();
  }

  // ---- epilogue: bias, convert, scatter bf16 ----
  const int h = wave;
#pragma unroll
  for (int tt = 0; tt < 4; ++tt) {
    const int d = tt * 16 + l15;
    const float bia = bias[h * 64 + d];
#pragma unroll
    for (int i = 0; i < 8; ++i) {
      const int r = row0 + i + half * 8;
      const int b = r >> 11;                  // rows per batch = 2048
      const int n = r & (N_ - 1);
      const float val = acc[tt][i] + bia;
      if (!transposed)
        out[((size_t)(b * H_ + h) * N_ + n) * DK_ + d] = f32_to_bf16(val);
      else
        out[((size_t)(b * H_ + h) * DV_ + d) * M_ + n] = f32_to_bf16(val);
    }
  }
}

// ---------------------------------------------------------------------------
// Kernel 3: flash attention. Block = 8 waves = 128 query rows of one (b,h).
// K (32x64) / V^T (64x32) chunks double-buffered in LDS via async-to-LDS DMA.
// ---------------------------------------------------------------------------
__global__ __launch_bounds__(256) void attn_kernel(
    const unsigned short* __restrict__ Qbf,   // [bh][n][dk] bf16
    const unsigned short* __restrict__ Kbf,   // [bh][m][dk] bf16
    const unsigned short* __restrict__ VTbf,  // [bh][dv][m] bf16
    float* __restrict__ outp) {               // [b][n][h][dv] fp32
  __shared__ alignas(16) unsigned short Klds[2][32 * 64];
  __shared__ alignas(16) unsigned short Vlds[2][64 * 32];
  __shared__ alignas(16) unsigned short Plds[8][16 * 32];

  const int lane = threadIdx.x & 31;
  const int wave = threadIdx.x >> 5;
  const int half = lane >> 4;
  const int l15  = lane & 15;
  const int bh   = blockIdx.x >> 4;
  const int nblk = blockIdx.x & 15;
  const int b = bh >> 3;
  const int h = bh & 7;
  const int n0 = nblk * 128 + wave * 16;

  // Q fragments for dk 0..31 / 32..63 (A layout, registers)
  FragBF qa0, qa1;
  {
    const unsigned short* qp = Qbf + ((size_t)bh * N_ + n0 + l15) * DK_;
    qa0.q[0] = *(const int4*)(qp + half * 8);
    qa0.q[1] = *(const int4*)(qp + half * 8 + 16);
    qa1.q[0] = *(const int4*)(qp + 32 + half * 8);
    qa1.q[1] = *(const int4*)(qp + 32 + half * 8 + 16);
  }

  v8f acc[4] = {};
  float mi[8], li[8];
#pragma unroll
  for (int i = 0; i < 8; ++i) { mi[i] = -1e30f; li[i] = 0.0f; }
  const float scale = 0.125f;                 // 1/sqrt(64)

  const unsigned short* Kbase = Kbf  + (size_t)bh * M_  * DK_;
  const unsigned short* Vbase = VTbf + (size_t)bh * DV_ * M_;
  const int rK = threadIdx.x >> 3, sK = threadIdx.x & 7;   // 32 rows x 8 seg
  const int rV = threadIdx.x >> 2, sV = threadIdx.x & 3;   // 64 rows x 4 seg

  auto stage = [&](int c0, int buf) {
    g2l_16(Kbase + (size_t)(c0 + rK) * DK_ + sK * 8, &Klds[buf][rK * 64 + sK * 8]);
    g2l_16(Vbase + (size_t)rV * M_ + c0 + sV * 8,    &Vlds[buf][rV * 32 + sV * 8]);
  };

  stage(0, 0);

  for (int it = 0; it < M_ / 32; ++it) {
    const int buf = it & 1;
    wait_async_lds();
    __syncthreads();
    if (it + 1 < M_ / 32) stage((it + 1) * 32, buf ^ 1);

    const unsigned short* KL = &Klds[buf][0];
    const unsigned short* VL = &Vlds[buf][0];

    // ---- load all 4 K B-fragments, then 4 WMMAs for S ----
    FragBF kb0, kb1, kb2, kb3;
    {
      const unsigned short* kp0 = KL + l15 * 64;
      const unsigned short* kp1 = KL + (16 + l15) * 64;
      kb0.q[0] = *(const int4*)(kp0 + half * 8);
      kb0.q[1] = *(const int4*)(kp0 + half * 8 + 16);
      kb1.q[0] = *(const int4*)(kp0 + 32 + half * 8);
      kb1.q[1] = *(const int4*)(kp0 + 32 + half * 8 + 16);
      kb2.q[0] = *(const int4*)(kp1 + half * 8);
      kb2.q[1] = *(const int4*)(kp1 + half * 8 + 16);
      kb3.q[0] = *(const int4*)(kp1 + 32 + half * 8);
      kb3.q[1] = *(const int4*)(kp1 + 32 + half * 8 + 16);
    }
    v8f s0 = {}, s1 = {};
    s0 = wmma_bf16(qa0, kb0, s0);
    s0 = wmma_bf16(qa1, kb1, s0);
    s1 = wmma_bf16(qa0, kb2, s1);
    s1 = wmma_bf16(qa1, kb3, s1);

    // ---- online softmax (rows i+8*half; reduce within 16-lane halves) ----
    float corr[8];
#pragma unroll
    for (int i = 0; i < 8; ++i) {
      float v0 = s0[i] * scale, v1 = s1[i] * scale;
      float cm = fmaxf(v0, v1);
#pragma unroll
      for (int msk = 1; msk < 16; msk <<= 1)
        cm = fmaxf(cm, __shfl_xor(cm, msk, 32));
      float mnew = fmaxf(mi[i], cm);
      corr[i] = __expf(mi[i] - mnew);
      float p0 = __expf(v0 - mnew);
      float p1 = __expf(v1 - mnew);
      float rs = p0 + p1;
#pragma unroll
      for (int msk = 1; msk < 16; msk <<= 1)
        rs += __shfl_xor(rs, msk, 32);
      li[i] = li[i] * corr[i] + rs;
      mi[i] = mnew;
      Plds[wave][(i + half * 8) * 32 + l15]      = f32_to_bf16(p0);
      Plds[wave][(i + half * 8) * 32 + 16 + l15] = f32_to_bf16(p1);
    }
#pragma unroll
    for (int t = 0; t < 4; ++t)
#pragma unroll
      for (int i = 0; i < 8; ++i) acc[t][i] *= corr[i];

    // ---- P fragment + all 4 V B-fragments, then 4 WMMAs for O += P V ----
    FragBF pa, vb0, vb1, vb2, vb3;
    {
      const unsigned short* pp = &Plds[wave][l15 * 32];
      pa.q[0] = *(const int4*)(pp + half * 16);
      pa.q[1] = *(const int4*)(pp + half * 16 + 8);
      const unsigned short* vp0 = VL + l15 * 32 + half * 16;
      vb0.q[0] = *(const int4*)(vp0);
      vb0.q[1] = *(const int4*)(vp0 + 8);
      vb1.q[0] = *(const int4*)(vp0 + 16 * 32);
      vb1.q[1] = *(const int4*)(vp0 + 16 * 32 + 8);
      vb2.q[0] = *(const int4*)(vp0 + 32 * 32);
      vb2.q[1] = *(const int4*)(vp0 + 32 * 32 + 8);
      vb3.q[0] = *(const int4*)(vp0 + 48 * 32);
      vb3.q[1] = *(const int4*)(vp0 + 48 * 32 + 8);
    }
    acc[0] = wmma_bf16(pa, vb0, acc[0]);
    acc[1] = wmma_bf16(pa, vb1, acc[1]);
    acc[2] = wmma_bf16(pa, vb2, acc[2]);
    acc[3] = wmma_bf16(pa, vb3, acc[3]);
  }

  // ---- epilogue: normalize and store fp32 out[b][n][h][dv] ----
#pragma unroll
  for (int t = 0; t < 4; ++t) {
    const int dv = t * 16 + l15;
#pragma unroll
    for (int i = 0; i < 8; ++i) {
      const int n = n0 + i + half * 8;
      outp[(((size_t)b * N_ + n) * H_ + h) * DV_ + dv] = acc[t][i] / li[i];
    }
  }
}

// ---------------------------------------------------------------------------
extern "C" void kernel_launch(void* const* d_in, const int* in_sizes, int n_in,
                              void* d_out, int out_size, void* d_ws,
                              size_t ws_size, hipStream_t stream) {
  (void)in_sizes; (void)n_in; (void)out_size; (void)ws_size;
  const float* first  = (const float*)d_in[0];
  const float* second = (const float*)d_in[1];
  const float* Wq = (const float*)d_in[2];
  const float* bq = (const float*)d_in[3];
  const float* Wk = (const float*)d_in[4];
  const float* bk = (const float*)d_in[5];
  const float* Wv = (const float*)d_in[6];
  const float* bv = (const float*)d_in[7];
  float* out = (float*)d_out;

  char* ws = (char*)d_ws;
  size_t off = 0;
  auto wsalloc = [&](size_t bytes) -> void* {
    void* p = ws + off;
    off += (bytes + 255) & ~(size_t)255;
    return p;
  };
  unsigned short* WTq = (unsigned short*)wsalloc((size_t)DIN_ * DIN_ * 2);
  unsigned short* WTk = (unsigned short*)wsalloc((size_t)DIN_ * DIN_ * 2);
  unsigned short* WTv = (unsigned short*)wsalloc((size_t)DIN_ * DIN_ * 2);
  unsigned short* Qb  = (unsigned short*)wsalloc((size_t)B_ * H_ * N_ * DK_ * 2);
  unsigned short* Kb  = (unsigned short*)wsalloc((size_t)B_ * H_ * M_ * DK_ * 2);
  unsigned short* Vt  = (unsigned short*)wsalloc((size_t)B_ * H_ * DV_ * M_ * 2);

  wt_cvt_kernel<<<3 * 16 * 16, 256, 0, stream>>>(Wq, Wk, Wv, WTq, WTk, WTv);
  proj_kernel<<<(B_ * N_) / 16, 256, 0, stream>>>(first,  WTq, bq, Qb, 0);
  proj_kernel<<<(B_ * M_) / 16, 256, 0, stream>>>(second, WTk, bk, Kb, 0);
  proj_kernel<<<(B_ * M_) / 16, 256, 0, stream>>>(second, WTv, bv, Vt, 1);
  attn_kernel<<<B_ * H_ * (N_ / 128), 256, 0, stream>>>(Qb, Kb, Vt, out);
}
